// Attention_3427383902235
// MI455X (gfx1250) — matrix operand order repaired and checked by reading, not compiled, not run on hardware
//
#include <hip/hip_runtime.h>

// ---------------------------------------------------------------------------
// Types for WMMA (CDNA5 / gfx1250, wave32)
// ---------------------------------------------------------------------------
typedef __attribute__((ext_vector_type(16))) __bf16 v16bf;
typedef __attribute__((ext_vector_type(8)))  float  v8f;

union Frag {
  uint4 u[2];
  v16bf v;
};
static_assert(sizeof(Frag) == 32, "frag size");

__device__ inline unsigned short f2bf(float f) {
  union { float f; unsigned int u; } c;
  c.f = f;
  unsigned int u = c.u;
  u += 0x7fffu + ((u >> 16) & 1u);   // round-to-nearest-even
  return (unsigned short)(u >> 16);
}

__device__ inline float redsum16(float x) {
  x += __shfl_xor(x, 1);
  x += __shfl_xor(x, 2);
  x += __shfl_xor(x, 4);
  x += __shfl_xor(x, 8);
  return x;
}
__device__ inline float redmax16(float x) {
  x = fmaxf(x, __shfl_xor(x, 1));
  x = fmaxf(x, __shfl_xor(x, 2));
  x = fmaxf(x, __shfl_xor(x, 4));
  x = fmaxf(x, __shfl_xor(x, 8));
  return x;
}

__device__ inline v8f wmma_bf16(const Frag& a, const Frag& b, v8f c) {
  return __builtin_amdgcn_wmma_f32_16x16x32_bf16(
      /*neg_a=*/false, a.v, /*neg_b=*/false, b.v,
      /*c_mod=*/(short)0, c, /*reuse_a=*/false, /*reuse_b=*/false);
}

// ---------------------------------------------------------------------------
// Problem constants
// ---------------------------------------------------------------------------
#define BATCH 4
#define SEQ   2048
#define DIM   768
#define HEADS 12
#define HDIM  64
#define TOKS  (BATCH * SEQ)   // 8192

// ---------------------------------------------------------------------------
// K0: fp32 -> bf16 (raw ushort) conversion, 4-wide
// ---------------------------------------------------------------------------
__global__ void cvt_f32_to_bf16_v4(const float4* __restrict__ in,
                                   ushort4* __restrict__ out, int n4) {
  int i = blockIdx.x * blockDim.x + threadIdx.x;
  if (i >= n4) return;
  float4 f = in[i];
  ushort4 r;
  r.x = f2bf(f.x);
  r.y = f2bf(f.y);
  r.z = f2bf(f.z);
  r.w = f2bf(f.w);
  out[i] = r;
}

// ---------------------------------------------------------------------------
// K1: qkv = x @ W^T + b, fused per-head LayerNorm on q,k, q pre-scaled.
//     grid = (TOKS/64, 2304/64), block = 128 (4 waves, 16 tokens each)
//     Q,K stored [bh, n, d] bf16; V stored transposed [bh, d, n] bf16.
// ---------------------------------------------------------------------------
__global__ __launch_bounds__(128) void qkv_ln_kernel(
    const unsigned short* __restrict__ xb,    // [8192, 768] bf16
    const unsigned short* __restrict__ wb,    // [2304, 768] bf16
    const float* __restrict__ bias,           // [2304]
    const float* __restrict__ qg, const float* __restrict__ qbeta,
    const float* __restrict__ kg, const float* __restrict__ kbeta,
    unsigned short* __restrict__ Qb,          // [48, 2048, 64]
    unsigned short* __restrict__ Kb,          // [48, 2048, 64]
    unsigned short* __restrict__ Vt)          // [48, 64, 2048]
{
  const int wave  = threadIdx.x >> 5;
  const int lane  = threadIdx.x & 31;
  const int lhalf = lane >> 4;
  const int l16   = lane & 15;
  const int tok0  = blockIdx.x * 64 + wave * 16;
  const int col0  = blockIdx.y * 64;

  v8f zero = {};
  v8f acc[4] = {zero, zero, zero, zero};

  const unsigned short* xrow = xb + (size_t)(tok0 + l16) * DIM;
  const unsigned short* wrow[4];
#pragma unroll
  for (int nb = 0; nb < 4; ++nb)
    wrow[nb] = wb + (size_t)(col0 + nb * 16 + l16) * DIM;

  for (int k0 = 0; k0 < DIM; k0 += 32) {
    Frag a;
    a.u[0] = *(const uint4*)(xrow + k0 + lhalf * 8);
    a.u[1] = *(const uint4*)(xrow + k0 + 16 + lhalf * 8);
#pragma unroll
    for (int nb = 0; nb < 4; ++nb) {
      Frag bm;
      bm.u[0] = *(const uint4*)(wrow[nb] + k0 + lhalf * 8);
      bm.u[1] = *(const uint4*)(wrow[nb] + k0 + 16 + lhalf * 8);
      acc[nb] = wmma_bf16(a, bm, acc[nb]);
    }
  }

  float bcol[4];
#pragma unroll
  for (int nb = 0; nb < 4; ++nb) bcol[nb] = bias[col0 + nb * 16 + l16];
#pragma unroll
  for (int nb = 0; nb < 4; ++nb)
#pragma unroll
    for (int v = 0; v < 8; ++v) acc[nb][v] += bcol[nb];

  const int which = blockIdx.y / 12;  // 0=q, 1=k, 2=v
  const int h     = blockIdx.y % 12;

  if (which < 2) {
    const float* g  = (which == 0) ? qg : kg;
    const float* bt = (which == 0) ? qbeta : kbeta;
    const float sc  = (which == 0) ? 0.125f : 1.0f;  // Dh^-0.5 folded into q
    float gv[4], bv[4];
#pragma unroll
    for (int nb = 0; nb < 4; ++nb) {
      gv[nb] = g[nb * 16 + l16];
      bv[nb] = bt[nb * 16 + l16];
    }
#pragma unroll
    for (int v = 0; v < 8; ++v) {
      float s1 = 0.0f, s2 = 0.0f;
#pragma unroll
      for (int nb = 0; nb < 4; ++nb) {
        float t = acc[nb][v];
        s1 += t;
        s2 += t * t;
      }
      s1 = redsum16(s1);
      s2 = redsum16(s2);
      float mu  = s1 * (1.0f / 64.0f);
      float var = s2 * (1.0f / 64.0f) - mu * mu;
      float r   = rsqrtf(var + 1e-5f);
#pragma unroll
      for (int nb = 0; nb < 4; ++nb)
        acc[nb][v] = ((acc[nb][v] - mu) * r * gv[nb] + bv[nb]) * sc;
    }
  }

  if (which == 2) {
    // V: packed transposed store, 8 consecutive tokens per lane
    const int t0 = tok0 + lhalf * 8;
    const int b  = t0 >> 11;
    const int n0 = t0 & (SEQ - 1);
    const size_t bh = (size_t)(b * HEADS + h);
#pragma unroll
    for (int nb = 0; nb < 4; ++nb) {
      union { unsigned short hh[8]; uint4 u; } pk;
#pragma unroll
      for (int v = 0; v < 8; ++v) pk.hh[v] = f2bf(acc[nb][v]);
      *(uint4*)(Vt + (bh * HDIM + nb * 16 + l16) * SEQ + n0) = pk.u;
    }
  } else {
    unsigned short* dst = (which == 0) ? Qb : Kb;
#pragma unroll
    for (int v = 0; v < 8; ++v) {
      const int t = tok0 + v + lhalf * 8;
      const int b = t >> 11;
      const int n = t & (SEQ - 1);
      const size_t base = ((size_t)(b * HEADS + h) * SEQ + n) * HDIM;
#pragma unroll
      for (int nb = 0; nb < 4; ++nb)
        dst[base + nb * 16 + l16] = f2bf(acc[nb][v]);
    }
  }
}

// ---------------------------------------------------------------------------
// K2: flash attention per (b,h). grid = (SEQ/64, B*H), block = 128.
//     Each wave: 16 queries, iterates keys in blocks of 64.
// ---------------------------------------------------------------------------
__global__ __launch_bounds__(128) void attn_kernel(
    const unsigned short* __restrict__ Qb,
    const unsigned short* __restrict__ Kb,
    const unsigned short* __restrict__ Vt,
    unsigned short* __restrict__ attnb)   // [8192, 768] bf16
{
  __shared__ unsigned short plds[4 * 16 * 64];  // 8 KB: P staging per wave
  const int wave  = threadIdx.x >> 5;
  const int lane  = threadIdx.x & 31;
  const int lhalf = lane >> 4;
  const int l16   = lane & 15;
  const int bh    = blockIdx.y;
  const int q0    = blockIdx.x * 64 + wave * 16;

  const unsigned short* Qh = Qb + (size_t)bh * (SEQ * HDIM);
  const unsigned short* Kh = Kb + (size_t)bh * (SEQ * HDIM);
  const unsigned short* Vh = Vt + (size_t)bh * (HDIM * SEQ);

  // Q A-fragments (Dh=64 -> two K=32 fragments), held for the whole loop
  Frag aq[2];
  const unsigned short* qrow = Qh + (size_t)(q0 + l16) * HDIM;
#pragma unroll
  for (int kb = 0; kb < 2; ++kb) {
    aq[kb].u[0] = *(const uint4*)(qrow + kb * 32 + lhalf * 8);
    aq[kb].u[1] = *(const uint4*)(qrow + kb * 32 + 16 + lhalf * 8);
  }

  v8f zero = {};
  v8f o[4] = {zero, zero, zero, zero};
  float m[8], l[8];
#pragma unroll
  for (int v = 0; v < 8; ++v) { m[v] = -3.0e38f; l[v] = 0.0f; }

  unsigned short* myp = plds + wave * (16 * 64);

  for (int key0 = 0; key0 < SEQ; key0 += 64) {
    if (key0 + 64 < SEQ) {
      __builtin_prefetch(Kh + (size_t)(key0 + 64 + l16) * HDIM, 0, 0);
      __builtin_prefetch(Vh + (size_t)l16 * SEQ + key0 + 64, 0, 0);
    }

    // S = (Q*scale) @ K^T : 4 key sub-tiles x 2 d-halves
    v8f s[4] = {zero, zero, zero, zero};
#pragma unroll
    for (int nb = 0; nb < 4; ++nb) {
      const unsigned short* krow = Kh + (size_t)(key0 + nb * 16 + l16) * HDIM;
#pragma unroll
      for (int kb = 0; kb < 2; ++kb) {
        Frag bk;
        bk.u[0] = *(const uint4*)(krow + kb * 32 + lhalf * 8);
        bk.u[1] = *(const uint4*)(krow + kb * 32 + 16 + lhalf * 8);
        s[nb] = wmma_bf16(aq[kb], bk, s[nb]);
      }
    }

    // online softmax over the 64 keys of this block
#pragma unroll
    for (int v = 0; v < 8; ++v) {
      float rm = fmaxf(fmaxf(s[0][v], s[1][v]), fmaxf(s[2][v], s[3][v]));
      rm = redmax16(rm);
      float nm   = fmaxf(m[v], rm);
      float corr = __expf(m[v] - nm);
      float rs   = 0.0f;
#pragma unroll
      for (int nb = 0; nb < 4; ++nb) {
        float p = __expf(s[nb][v] - nm);
        s[nb][v] = p;
        rs += p;
      }
      rs   = redsum16(rs);
      l[v] = l[v] * corr + rs;
      m[v] = nm;
#pragma unroll
      for (int nb = 0; nb < 4; ++nb) o[nb][v] *= corr;
    }

    // remap P from C-layout to A-fragment layout via per-wave LDS patch
#pragma unroll
    for (int nb = 0; nb < 4; ++nb)
#pragma unroll
      for (int v = 0; v < 8; ++v)
        myp[(v + 8 * lhalf) * 64 + nb * 16 + l16] = f2bf(s[nb][v]);

    // per-wave RAW through LDS: force stores to complete before reads
    asm volatile("s_wait_dscnt 0" ::: "memory");

    Frag ap[2];
    const unsigned short* prow = myp + l16 * 64;
#pragma unroll
    for (int kb = 0; kb < 2; ++kb) {
      ap[kb].u[0] = *(const uint4*)(prow + kb * 32 + lhalf * 8);
      ap[kb].u[1] = *(const uint4*)(prow + kb * 32 + 16 + lhalf * 8);
    }

    // O += P @ V  (V transposed: contiguous per-lane fragment loads)
#pragma unroll
    for (int nb = 0; nb < 4; ++nb) {
      const unsigned short* vrow = Vh + (size_t)(nb * 16 + l16) * SEQ + key0;
#pragma unroll
      for (int kb = 0; kb < 2; ++kb) {
        Frag bv;
        bv.u[0] = *(const uint4*)(vrow + kb * 32 + lhalf * 8);
        bv.u[1] = *(const uint4*)(vrow + kb * 32 + 16 + lhalf * 8);
        o[nb] = wmma_bf16(ap[kb], bv, o[nb]);
      }
    }
  }

  // finalize: divide by l, store [b, n, h*64+d] bf16
  const int h = bh % HEADS;
  const int b = bh / HEADS;
#pragma unroll
  for (int v = 0; v < 8; ++v) {
    const float inv = 1.0f / l[v];
    const int q = q0 + v + lhalf * 8;
    const size_t base = ((size_t)(b * SEQ + q)) * DIM + h * HDIM;
#pragma unroll
    for (int nb = 0; nb < 4; ++nb)
      attnb[base + nb * 16 + l16] = f2bf(o[nb][v] * inv);
  }
}

// ---------------------------------------------------------------------------
// K3: out = attn @ proj_w^T + proj_b, fp32 output.
//     grid = (TOKS/64, DIM/64), block = 128.
// ---------------------------------------------------------------------------
__global__ __launch_bounds__(128) void proj_kernel(
    const unsigned short* __restrict__ ab,   // [8192, 768] bf16
    const unsigned short* __restrict__ wb,   // [768, 768] bf16
    const float* __restrict__ bias,          // [768]
    float* __restrict__ out)                 // [8192, 768] f32
{
  const int wave  = threadIdx.x >> 5;
  const int lane  = threadIdx.x & 31;
  const int lhalf = lane >> 4;
  const int l16   = lane & 15;
  const int tok0  = blockIdx.x * 64 + wave * 16;
  const int col0  = blockIdx.y * 64;

  v8f zero = {};
  v8f acc[4] = {zero, zero, zero, zero};

  const unsigned short* arow = ab + (size_t)(tok0 + l16) * DIM;
  const unsigned short* wrow[4];
#pragma unroll
  for (int nb = 0; nb < 4; ++nb)
    wrow[nb] = wb + (size_t)(col0 + nb * 16 + l16) * DIM;

  for (int k0 = 0; k0 < DIM; k0 += 32) {
    Frag a;
    a.u[0] = *(const uint4*)(arow + k0 + lhalf * 8);
    a.u[1] = *(const uint4*)(arow + k0 + 16 + lhalf * 8);
#pragma unroll
    for (int nb = 0; nb < 4; ++nb) {
      Frag bm;
      bm.u[0] = *(const uint4*)(wrow[nb] + k0 + lhalf * 8);
      bm.u[1] = *(const uint4*)(wrow[nb] + k0 + 16 + lhalf * 8);
      acc[nb] = wmma_bf16(a, bm, acc[nb]);
    }
  }

  float bcol[4];
#pragma unroll
  for (int nb = 0; nb < 4; ++nb) bcol[nb] = bias[col0 + nb * 16 + l16];

#pragma unroll
  for (int v = 0; v < 8; ++v) {
    const int t = tok0 + v + lhalf * 8;
#pragma unroll
    for (int nb = 0; nb < 4; ++nb)
      out[(size_t)t * DIM + col0 + nb * 16 + l16] = acc[nb][v] + bcol[nb];
  }
}

// ---------------------------------------------------------------------------
// Host launcher
// ---------------------------------------------------------------------------
extern "C" void kernel_launch(void* const* d_in, const int* in_sizes, int n_in,
                              void* d_out, int out_size, void* d_ws, size_t ws_size,
                              hipStream_t stream) {
  (void)in_sizes; (void)n_in; (void)out_size; (void)ws_size;

  const float* x      = (const float*)d_in[0];
  const float* qkv_w  = (const float*)d_in[1];
  const float* qkv_b  = (const float*)d_in[2];
  const float* proj_w = (const float*)d_in[3];
  const float* proj_b = (const float*)d_in[4];
  const float* q_g    = (const float*)d_in[5];
  const float* q_b    = (const float*)d_in[6];
  const float* k_g    = (const float*)d_in[7];
  const float* k_b    = (const float*)d_in[8];
  float* out = (float*)d_out;

  unsigned short* ws = (unsigned short*)d_ws;
  unsigned short* xb    = ws; ws += (size_t)TOKS * DIM;          // 12.6 MB
  unsigned short* wqkv  = ws; ws += (size_t)3 * DIM * DIM;       //  3.5 MB
  unsigned short* wproj = ws; ws += (size_t)DIM * DIM;           //  1.2 MB
  unsigned short* Qb    = ws; ws += (size_t)BATCH * HEADS * SEQ * HDIM;
  unsigned short* Kb    = ws; ws += (size_t)BATCH * HEADS * SEQ * HDIM;
  unsigned short* Vt    = ws; ws += (size_t)BATCH * HEADS * HDIM * SEQ;
  unsigned short* attnb = ws; ws += (size_t)TOKS * DIM;

  {
    int n4 = TOKS * DIM / 4;
    cvt_f32_to_bf16_v4<<<(n4 + 255) / 256, 256, 0, stream>>>(
        (const float4*)x, (ushort4*)xb, n4);
  }
  {
    int n4 = 3 * DIM * DIM / 4;
    cvt_f32_to_bf16_v4<<<(n4 + 255) / 256, 256, 0, stream>>>(
        (const float4*)qkv_w, (ushort4*)wqkv, n4);
  }
  {
    int n4 = DIM * DIM / 4;
    cvt_f32_to_bf16_v4<<<(n4 + 255) / 256, 256, 0, stream>>>(
        (const float4*)proj_w, (ushort4*)wproj, n4);
  }

  qkv_ln_kernel<<<dim3(TOKS / 64, (3 * DIM) / 64), 128, 0, stream>>>(
      xb, wqkv, qkv_b, q_g, q_b, k_g, k_b, Qb, Kb, Vt);

  attn_kernel<<<dim3(SEQ / 64, BATCH * HEADS), 128, 0, stream>>>(
      Qb, Kb, Vt, attnb);

  proj_kernel<<<dim3(TOKS / 64, DIM / 64), 128, 0, stream>>>(
      attnb, wproj, proj_b, out);
}
